// TransformerBlock_59588376265154
// MI455X (gfx1250) — matrix operand order repaired and checked by reading, not compiled, not run on hardware
//
#include <hip/hip_runtime.h>

#define DEV __device__ __forceinline__

// ---- types --------------------------------------------------------------
typedef __attribute__((ext_vector_type(16))) __bf16        v16bf;
typedef __attribute__((ext_vector_type(8)))  float         v8f;
typedef __attribute__((ext_vector_type(4)))  unsigned int  u32x4;
typedef __attribute__((ext_vector_type(2)))  unsigned int  u32x2;

union FragB16 {                 // one 16x32 (A) or 32x16 (B) bf16 WMMA operand
  v16bf          v;
  unsigned short h[16];
  u32x2          u64[4];
  u32x4          u128[2];
};

static DEV unsigned short f2bf(float f) {      // round-to-nearest-even
  unsigned int u = __float_as_uint(f);
  u += 0x7FFFu + ((u >> 16) & 1u);
  return (unsigned short)(u >> 16);
}

static DEV v8f vzero8() { v8f z = {0.f,0.f,0.f,0.f,0.f,0.f,0.f,0.f}; return z; }

#define WMMA_BF16(A, B, C) \
  __builtin_amdgcn_wmma_f32_16x16x32_bf16(false, (A), false, (B), (short)0, (C), false, false)

// CDNA5 async copy: per-lane 16B global -> LDS, tracked by ASYNCcnt.
// vdst = LDS byte address (low 32 bits of flat shared pointer), vaddr = 64-bit global.
#define ASYNC_LOAD_B128(ldsaddr, gptr) \
  asm volatile("global_load_async_to_lds_b128 %0, %1, off" \
               :: "v"(ldsaddr), "v"(gptr) : "memory")
#define ASYNC_WAIT0() asm volatile("s_wait_asynccnt 0x0" ::: "memory")

static DEV unsigned lds_off(const void* p) { return (unsigned)(size_t)p; }

// ---- problem constants --------------------------------------------------
#define BB    4
#define TT    2048
#define CC    1024
#define HH    16
#define HSZ   64
#define C3    3072
#define FFD   4096
#define MROWS (BB * TT)

// ========================================================================
// Weight prep: pack Wq/Wk/Wv ([H,C,HS] f32) -> bf16 [3C][C] (K-major)
// ========================================================================
__global__ __launch_bounds__(256)
void pack_wqkv(const float* __restrict__ Wq, const float* __restrict__ Wk,
               const float* __restrict__ Wv, unsigned short* __restrict__ out)
{
  size_t idx = (size_t)blockIdx.x * 256 + threadIdx.x;   // over 3C*C
  int c  = (int)(idx & (CC - 1));
  int n  = (int)(idx >> 10);
  const float* W = (n < CC) ? Wq : (n < 2 * CC ? Wk : Wv);
  int nn = n & (CC - 1);
  int h  = nn >> 6;
  int d  = nn & (HSZ - 1);
  out[idx] = f2bf(W[((size_t)h * CC + c) * HSZ + d]);
}

// in: [K][N] f32 row-major  ->  out: [N][K] bf16 (K contiguous)
__global__ __launch_bounds__(256)
void transpose_to_bf16(const float* __restrict__ in, unsigned short* __restrict__ out,
                       int K, int N)
{
  size_t idx = (size_t)blockIdx.x * 256 + threadIdx.x;   // over N*K
  int    k   = (int)(idx % K);
  size_t n   = idx / K;
  out[idx]   = f2bf(in[(size_t)k * N + n]);
}

// ========================================================================
// LayerNorm (C=1024) -> bf16 activations
// ========================================================================
__global__ __launch_bounds__(256)
void layernorm_to_bf16(const float* __restrict__ x, const float* __restrict__ g,
                       const float* __restrict__ be, unsigned short* __restrict__ out)
{
  const int row = blockIdx.x;
  const float* xr = x + (size_t)row * CC;
  const int t = threadIdx.x;
  float v[4], s = 0.f, ss = 0.f;
  #pragma unroll
  for (int i = 0; i < 4; ++i) { v[i] = xr[t + i * 256]; s += v[i]; ss += v[i] * v[i]; }
  #pragma unroll
  for (int m = 16; m; m >>= 1) { s += __shfl_xor(s, m, 32); ss += __shfl_xor(ss, m, 32); }
  __shared__ float rs[8], rss[8];
  if ((t & 31) == 0) { rs[t >> 5] = s; rss[t >> 5] = ss; }
  __syncthreads();
  s = 0.f; ss = 0.f;
  #pragma unroll
  for (int w = 0; w < 8; ++w) { s += rs[w]; ss += rss[w]; }
  const float mu   = s  * (1.f / CC);
  const float var  = ss * (1.f / CC) - mu * mu;
  const float rinv = rsqrtf(var + 1e-5f);
  #pragma unroll
  for (int i = 0; i < 4; ++i) {
    const int c = t + i * 256;
    out[(size_t)row * CC + c] = f2bf((v[i] - mu) * rinv * g[c] + be[c]);
  }
}

// ========================================================================
// bf16 WMMA GEMM:  out = epilogue(A[M,K] * Bt[N,K]^T)
//   MODE 0: f32 out = acc + bias + residual
//   MODE 1: bf16 out = acc
//   MODE 2: bf16 out = relu(acc + bias)
// Block tile 128(M) x 256(N), 8 waves of 64x64, K-step 32.
// Double-buffered LDS filled by global_load_async_to_lds_b128 (ASYNCcnt).
// ========================================================================
template <int MODE>
__global__ __launch_bounds__(256)
void gemm_bf16_wmma(const unsigned short* __restrict__ A,
                    const unsigned short* __restrict__ Bt,
                    const float* __restrict__ bias,
                    const float* __restrict__ resid,
                    void* __restrict__ out,
                    int M, int N, int K)
{
  // rows 0..127 = A tile (128x32), rows 128..383 = B tile (256x32); padded rows
  __shared__ unsigned short Ts[2][384][40];

  const int tid  = threadIdx.x;
  const int lane = tid & 31;
  const int wid  = tid >> 5;
  const int wm   = wid & 1;          // 2 row groups of 64
  const int wn   = wid >> 1;         // 4 col groups of 64
  const int m0   = blockIdx.y * 128;
  const int n0   = blockIdx.x * 256;
  const int ml   = lane & 15;
  const int hf   = lane >> 4;

  // async-stage one 384x32 tile: 256 threads x 3 row-halves x 2 x b128
  auto stage = [&](int k0, int b) {
    #pragma unroll
    for (int r = 0; r < 3; ++r) {
      const int rh  = tid + r * 256;       // row-half id
      const int row = rh >> 1;
      const int lk  = (rh & 1) * 16;
      const unsigned short* gsrc = (row < 128)
          ? (A  + (size_t)(m0 + row)       * K + k0 + lk)
          : (Bt + (size_t)(n0 + row - 128) * K + k0 + lk);
      __builtin_prefetch((const void*)(gsrc + 32), 0, 3);   // tile k0+32 ahead
      ASYNC_LOAD_B128(lds_off(&Ts[b][row][lk]),     gsrc);
      ASYNC_LOAD_B128(lds_off(&Ts[b][row][lk + 8]), gsrc + 8);
    }
  };

  v8f acc[4][4];
  #pragma unroll
  for (int i = 0; i < 4; ++i)
    #pragma unroll
    for (int j = 0; j < 4; ++j) acc[i][j] = vzero8();

  stage(0, 0);
  ASYNC_WAIT0();
  __syncthreads();

  int buf = 0;
  for (int k0 = 0; k0 < K; k0 += 32) {
    const bool hasNext = (k0 + 32) < K;
    if (hasNext) stage(k0 + 32, buf ^ 1);   // overlaps with compute below

    FragB16 af[4];
    #pragma unroll
    for (int i = 0; i < 4; ++i) {   // A 16x32: e<8 -> k=e+hf*8 ; e>=8 -> 16+hf*8+(e-8)
      const int ra = wm * 64 + i * 16 + ml;
      af[i].u128[0] = *(const u32x4*)&Ts[buf][ra][hf * 8];
      af[i].u128[1] = *(const u32x4*)&Ts[buf][ra][16 + hf * 8];
    }
    #pragma unroll
    for (int j = 0; j < 4; ++j) {   // B 32x16: e<8 -> k=e+hf*8 ; e>=8 -> 16+hf*8+(e-8), column n = lane&15
      const int rb = 128 + wn * 64 + j * 16 + ml;
      FragB16 bfm;
      bfm.u128[0] = *(const u32x4*)&Ts[buf][rb][hf * 8];
      bfm.u128[1] = *(const u32x4*)&Ts[buf][rb][16 + hf * 8];
      #pragma unroll
      for (int i = 0; i < 4; ++i)
        acc[i][j] = WMMA_BF16(af[i].v, bfm.v, acc[i][j]);
    }

    if (hasNext) ASYNC_WAIT0();
    __syncthreads();
    buf ^= 1;
  }

  // ---- branch-free epilogue (C layout: M = r + 8*hf, N = lane&15) ----
  float bj[4];
  #pragma unroll
  for (int j = 0; j < 4; ++j)
    bj[j] = (MODE != 1) ? bias[n0 + wn * 64 + j * 16 + ml] : 0.f;

  #pragma unroll
  for (int i = 0; i < 4; ++i)
    #pragma unroll
    for (int r = 0; r < 8; ++r) {
      const int    gm   = m0 + wm * 64 + i * 16 + r + hf * 8;
      const size_t base = (size_t)gm * N + n0 + wn * 64 + ml;
      #pragma unroll
      for (int j = 0; j < 4; ++j) {
        const size_t o = base + j * 16;
        float v = acc[i][j][r];
        if (MODE != 1) v += bj[j];
        if (MODE == 0) {
          ((float*)out)[o] = v + resid[o];
        } else {
          if (MODE == 2) v = fmaxf(v, 0.f);
          ((unsigned short*)out)[o] = f2bf(v);
        }
      }
    }
}

// ========================================================================
// Flash attention (causal), bf16 WMMA, f32 online softmax.
// grid = (T/128, B*H); block = 256 (8 waves x 16 query rows).
// K/V chunks double-buffered via async global->LDS copies.
// ========================================================================
__global__ __launch_bounds__(256)
void attention_wmma(const unsigned short* __restrict__ qkv,
                    unsigned short* __restrict__ attnOut)
{
  __shared__ unsigned short Ks[2][32][80];    // 32 keys x 64 dims (padded)
  __shared__ unsigned short Vs[2][32][80];
  __shared__ unsigned short Ps[8][16][40];    // per-wave P relayout buffer

  const int tid  = threadIdx.x;
  const int lane = tid & 31;
  const int wid  = tid >> 5;
  const int ml   = lane & 15;
  const int hf   = lane >> 4;

  const int bh = blockIdx.y;
  const int b  = bh >> 4;
  const int h  = bh & 15;
  const int qbase = blockIdx.x * 128;
  const int qr0   = qbase + wid * 16;

  const size_t rowbase = (size_t)b * TT * C3;
  const int qcol = h * HSZ;
  const int kcol = CC + h * HSZ;
  const int vcol = 2 * CC + h * HSZ;

  // async-stage one 32-key K/V chunk (256 threads x 2 x b128)
  auto stageKV = [&](int s0, int bsel) {
    const int srow = tid >> 3;
    const int doff = (tid & 7) * 8;
    const size_t rb = rowbase + (size_t)(s0 + srow) * C3;
    ASYNC_LOAD_B128(lds_off(&Ks[bsel][srow][doff]), qkv + rb + kcol + doff);
    ASYNC_LOAD_B128(lds_off(&Vs[bsel][srow][doff]), qkv + rb + vcol + doff);
  };

  // Q fragments (A layout over d): i selects d-chunk {0..31, 32..63}
  FragB16 aq[2];
  #pragma unroll
  for (int i = 0; i < 2; ++i) {
    const unsigned short* qp =
        qkv + rowbase + (size_t)(qr0 + ml) * C3 + qcol + i * 32 + hf * 8;
    aq[i].u128[0] = *(const u32x4*)qp;
    aq[i].u128[1] = *(const u32x4*)(qp + 16);
  }

  v8f accO[4];
  #pragma unroll
  for (int j = 0; j < 4; ++j) accO[j] = vzero8();
  float mprev[8], lsum[8];
  #pragma unroll
  for (int r = 0; r < 8; ++r) { mprev[r] = -__builtin_inff(); lsum[r] = 0.f; }

  const float scale = 0.125f;   // 1/sqrt(64)
  const int   send  = qbase + 128;

  stageKV(0, 0);
  ASYNC_WAIT0();
  __syncthreads();

  int kb = 0;
  for (int s0 = 0; s0 < send; s0 += 32) {
    const bool hasNext = (s0 + 32) < send;
    if (hasNext) stageKV(s0 + 32, kb ^ 1);    // overlaps with compute

    if (s0 <= qr0 + 15) {                     // wave-uniform causal skip
      // ---- scores: S[16q x 32s] via 4 WMMAs ----
      v8f sc[2];
      #pragma unroll
      for (int js = 0; js < 2; ++js) {
        sc[js] = vzero8();
        const int scol = js * 16 + ml;        // B column = key index
        #pragma unroll
        for (int i = 0; i < 2; ++i) {         // d chunks
          FragB16 bk;
          bk.u128[0] = *(const u32x4*)&Ks[kb][scol][i * 32 + hf * 8];
          bk.u128[1] = *(const u32x4*)&Ks[kb][scol][i * 32 + 16 + hf * 8];
          sc[js] = WMMA_BF16(aq[i].v, bk.v, sc[js]);
        }
      }
      // ---- causal mask + online softmax (rows live in 16-lane halves) ----
      #pragma unroll
      for (int r = 0; r < 8; ++r) {
        const int qg = qr0 + r + hf * 8;
        float v0 = sc[0][r] * scale;
        float v1 = sc[1][r] * scale;
        if (s0 + ml > qg)      v0 = -__builtin_inff();
        if (s0 + 16 + ml > qg) v1 = -__builtin_inff();
        float cm = fmaxf(v0, v1);
        #pragma unroll
        for (int mk = 8; mk; mk >>= 1) cm = fmaxf(cm, __shfl_xor(cm, mk, 32));
        const float mnew  = fmaxf(mprev[r], cm);
        const float alpha = __expf(mprev[r] - mnew);
        const float p0 = __expf(v0 - mnew);
        const float p1 = __expf(v1 - mnew);
        float psum = p0 + p1;
        #pragma unroll
        for (int mk = 8; mk; mk >>= 1) psum += __shfl_xor(psum, mk, 32);
        lsum[r]  = lsum[r] * alpha + psum;
        mprev[r] = mnew;
        #pragma unroll
        for (int j = 0; j < 4; ++j) accO[j][r] *= alpha;
        Ps[wid][r + (hf << 3)][ml]      = f2bf(p0);
        Ps[wid][r + (hf << 3)][16 + ml] = f2bf(p1);
      }
      // ---- P (A layout) from per-wave LDS ----
      FragB16 ap;
      ap.u128[0] = *(const u32x4*)&Ps[wid][ml][hf * 8];
      ap.u128[1] = *(const u32x4*)&Ps[wid][ml][16 + hf * 8];
      // ---- O += P @ V : 4 WMMAs over d chunks ----
      #pragma unroll
      for (int j = 0; j < 4; ++j) {
        FragB16 bv;                            // B[k=s][n=d]
        #pragma unroll
        for (int e = 0; e < 16; ++e)
          bv.h[e] = Vs[kb][(e < 8 ? e : e + 8) + hf * 8][j * 16 + ml];
        accO[j] = WMMA_BF16(ap.v, bv.v, accO[j]);
      }
    }

    if (hasNext) ASYNC_WAIT0();
    __syncthreads();
    kb ^= 1;
  }

  // ---- normalize and write concat-head bf16 [B*T][C] ----
  #pragma unroll
  for (int j = 0; j < 4; ++j)
    #pragma unroll
    for (int r = 0; r < 8; ++r) {
      const int qg = qr0 + r + hf * 8;
      const float inv = 1.f / lsum[r];
      attnOut[((size_t)b * TT + qg) * CC + h * HSZ + j * 16 + ml] =
          f2bf(accO[j][r] * inv);
    }
}

// ========================================================================
// Host-side orchestration
// ========================================================================
extern "C" void kernel_launch(void* const* d_in, const int* in_sizes, int n_in,
                              void* d_out, int out_size, void* d_ws, size_t ws_size,
                              hipStream_t stream) {
  (void)in_sizes; (void)n_in; (void)out_size; (void)ws_size;

  const float* x     = (const float*)d_in[0];
  const float* Wq    = (const float*)d_in[1];
  const float* Wk    = (const float*)d_in[2];
  const float* Wv    = (const float*)d_in[3];
  const float* Wproj = (const float*)d_in[4];
  const float* bproj = (const float*)d_in[5];
  const float* W1    = (const float*)d_in[6];
  const float* b1    = (const float*)d_in[7];
  const float* W2    = (const float*)d_in[8];
  const float* b2    = (const float*)d_in[9];
  const float* g1    = (const float*)d_in[10];
  const float* be1   = (const float*)d_in[11];
  const float* g2    = (const float*)d_in[12];
  const float* be2   = (const float*)d_in[13];

  char* p = (char*)d_ws;
  auto take = [&](size_t elems, size_t esize) -> char* {
    char* r = p;
    p += (elems * esize + 255) & ~(size_t)255;
    return r;
  };
  unsigned short* h1     = (unsigned short*)take((size_t)MROWS * CC,  2);
  unsigned short* qkv    = (unsigned short*)take((size_t)MROWS * C3,  2);
  unsigned short* attnO  = (unsigned short*)take((size_t)MROWS * CC,  2);
  float*          x1     = (float*)         take((size_t)MROWS * CC,  4);
  unsigned short* h2     = (unsigned short*)take((size_t)MROWS * CC,  2);
  unsigned short* ffh    = (unsigned short*)take((size_t)MROWS * FFD, 2);
  unsigned short* WqkvT  = (unsigned short*)take((size_t)C3 * CC,     2);
  unsigned short* WprojT = (unsigned short*)take((size_t)CC * CC,     2);
  unsigned short* W1T    = (unsigned short*)take((size_t)FFD * CC,    2);
  unsigned short* W2T    = (unsigned short*)take((size_t)CC * FFD,    2);

  // weight prep (bf16, K-major)
  pack_wqkv<<<(C3 * CC) / 256, 256, 0, stream>>>(Wq, Wk, Wv, WqkvT);
  transpose_to_bf16<<<(CC * CC)  / 256, 256, 0, stream>>>(Wproj, WprojT, CC,  CC);
  transpose_to_bf16<<<(FFD * CC) / 256, 256, 0, stream>>>(W1,    W1T,    CC,  FFD);
  transpose_to_bf16<<<(CC * FFD) / 256, 256, 0, stream>>>(W2,    W2T,    FFD, CC);

  // ln1 -> QKV -> attention -> proj(+x) -> ln2 -> FF1(relu) -> FF2(+x1)
  layernorm_to_bf16<<<MROWS, 256, 0, stream>>>(x, g1, be1, h1);
  gemm_bf16_wmma<1><<<dim3(C3 / 256, MROWS / 128), 256, 0, stream>>>(
      h1, WqkvT, nullptr, nullptr, qkv, MROWS, C3, CC);
  attention_wmma<<<dim3(TT / 128, BB * HH), 256, 0, stream>>>(qkv, attnO);
  gemm_bf16_wmma<0><<<dim3(CC / 256, MROWS / 128), 256, 0, stream>>>(
      attnO, WprojT, bproj, x, x1, MROWS, CC, CC);
  layernorm_to_bf16<<<MROWS, 256, 0, stream>>>(x1, g2, be2, h2);
  gemm_bf16_wmma<2><<<dim3(FFD / 256, MROWS / 128), 256, 0, stream>>>(
      h2, W1T, b1, nullptr, ffh, MROWS, FFD, CC);
  gemm_bf16_wmma<0><<<dim3(CC / 256, MROWS / 128), 256, 0, stream>>>(
      ffh, W2T, b2, x1, (float*)d_out, MROWS, CC, FFD);
}
